// Cigin_15058155340158
// MI455X (gfx1250) — matrix operand order repaired and checked by reading, not compiled, not run on hardware
//
#include <hip/hip_runtime.h>
#include <hip/hip_bf16.h>

// ---------- types for CDNA5 WMMA ----------
typedef __attribute__((ext_vector_type(16))) _Float16 v16h;
typedef __attribute__((ext_vector_type(8)))  _Float16 v8h;
typedef __attribute__((ext_vector_type(8)))  float    v8f;

#define N_NODES 8000
#define N_EDGES 128000
#define ND 40
#define ED 10
#define FD 80   // 2*ND (set2set output width)
#define KP 96   // K padded to 3 chunks of 32

__device__ __forceinline__ float sigm(float x) { return 1.f / (1.f + __expf(-x)); }

__device__ __forceinline__ v16h cat8(v8h lo, v8h hi) {
  return __builtin_shufflevector(lo, hi, 0, 1, 2, 3, 4, 5, 6, 7,
                                 8, 9, 10, 11, 12, 13, 14, 15);
}

// =====================================================================
// Message passing: per-edge message  msg_e = Uw @ [h[src], h[dst], ef] + Ub
// accumulated into msg[src] via f32 atomics (segment_sum over src).
// =====================================================================
__global__ void mp_edge(const float* __restrict__ h, const int* __restrict__ ei,
                        const float* __restrict__ ef,
                        const float* __restrict__ Uw, const float* __restrict__ Ub,
                        float* __restrict__ msg, int nE) {
  __shared__ float sW[ND * 90];
  __shared__ float sB[ND];
  for (int i = threadIdx.x; i < ND * 90; i += blockDim.x) sW[i] = Uw[i];
  for (int i = threadIdx.x; i < ND; i += blockDim.x) sB[i] = Ub[i];
  __syncthreads();
  int e = blockIdx.x * blockDim.x + threadIdx.x;
  if (e >= nE) return;
  int src = ei[e];
  int dst = ei[nE + e];
  float out[ND];
#pragma unroll
  for (int j = 0; j < ND; ++j) out[j] = sB[j];
  for (int c = 0; c < ND; ++c) {
    float z = h[src * ND + c];
#pragma unroll 8
    for (int j = 0; j < ND; ++j) out[j] += sW[j * 90 + c] * z;
  }
  for (int c = 0; c < ND; ++c) {
    float z = h[dst * ND + c];
#pragma unroll 8
    for (int j = 0; j < ND; ++j) out[j] += sW[j * 90 + 40 + c] * z;
  }
  for (int c = 0; c < ED; ++c) {
    float z = ef[e * ED + c];
#pragma unroll 8
    for (int j = 0; j < ND; ++j) out[j] += sW[j * 90 + 80 + c] * z;
  }
#pragma unroll
  for (int j = 0; j < ND; ++j) atomicAdd(&msg[src * ND + j], out[j]);
}

// node update: h = relu(Mw @ [h, msg] + Mb)
__global__ void mp_node(const float* __restrict__ h, const float* __restrict__ msg,
                        const float* __restrict__ Mw, const float* __restrict__ Mb,
                        float* __restrict__ hout, int n) {
  __shared__ float sW[ND * 80];
  __shared__ float sB[ND];
  for (int i = threadIdx.x; i < ND * 80; i += blockDim.x) sW[i] = Mw[i];
  for (int i = threadIdx.x; i < ND; i += blockDim.x) sB[i] = Mb[i];
  __syncthreads();
  int i = blockIdx.x * blockDim.x + threadIdx.x;
  if (i >= n) return;
  float out[ND];
#pragma unroll
  for (int j = 0; j < ND; ++j) out[j] = sB[j];
  for (int c = 0; c < ND; ++c) {
    float z = h[i * ND + c];
#pragma unroll 8
    for (int j = 0; j < ND; ++j) out[j] += sW[j * 80 + c] * z;
  }
  for (int c = 0; c < ND; ++c) {
    float z = msg[i * ND + c];
#pragma unroll 8
    for (int j = 0; j < ND; ++j) out[j] += sW[j * 80 + 40 + c] * z;
  }
  for (int j = 0; j < ND; ++j) hout[i * ND + j] = fmaxf(out[j], 0.f);
}

// =====================================================================
// set2set over the 2-element set {h0, hT}, one thread per node.
// q_star is represented as (h, r); q_star[0:40] == h always.
// =====================================================================
__global__ void s2s_pair(const float* __restrict__ h0s, const float* __restrict__ hts,
                         const float* __restrict__ Wih, const float* __restrict__ Whh,
                         const float* __restrict__ bih, const float* __restrict__ bhh,
                         float* __restrict__ qout, int n) {
  int i = blockIdx.x * blockDim.x + threadIdx.x;
  if (i >= n) return;
  float h0[ND], ht[ND], h[ND], c[ND], r[ND], nh[ND];
  for (int u = 0; u < ND; ++u) {
    h0[u] = h0s[i * ND + u];
    ht[u] = hts[i * ND + u];
    h[u] = 0.f; c[u] = 0.f; r[u] = 0.f;
  }
  for (int step = 0; step < 2; ++step) {
    for (int u = 0; u < ND; ++u) {
      float gi = bih[u] + bhh[u];
      float gf = bih[40 + u] + bhh[40 + u];
      float gg = bih[80 + u] + bhh[80 + u];
      float go = bih[120 + u] + bhh[120 + u];
      const float* wi0 = &Wih[u * 80];
      const float* wi1 = &Wih[(40 + u) * 80];
      const float* wi2 = &Wih[(80 + u) * 80];
      const float* wi3 = &Wih[(120 + u) * 80];
      const float* wh0 = &Whh[u * ND];
      const float* wh1 = &Whh[(40 + u) * ND];
      const float* wh2 = &Whh[(80 + u) * ND];
      const float* wh3 = &Whh[(120 + u) * ND];
      for (int c1 = 0; c1 < ND; ++c1) {
        float hq = h[c1], rq = r[c1];
        gi += wi0[c1] * hq + wi0[40 + c1] * rq + wh0[c1] * hq;
        gf += wi1[c1] * hq + wi1[40 + c1] * rq + wh1[c1] * hq;
        gg += wi2[c1] * hq + wi2[40 + c1] * rq + wh2[c1] * hq;
        go += wi3[c1] * hq + wi3[40 + c1] * rq + wh3[c1] * hq;
      }
      float cn = sigm(gf) * c[u] + sigm(gi) * tanhf(gg);
      c[u] = cn;
      nh[u] = sigm(go) * tanhf(cn);
    }
    float e0 = 0.f, e1 = 0.f;
    for (int u = 0; u < ND; ++u) { h[u] = nh[u]; e0 += h0[u] * h[u]; e1 += ht[u] * h[u]; }
    float m = fmaxf(e0, e1);
    float a0 = __expf(e0 - m), a1 = __expf(e1 - m);
    float s = a0 + a1;
    for (int u = 0; u < ND; ++u) r[u] = (a0 * h0[u] + a1 * ht[u]) / s;
  }
  for (int u = 0; u < ND; ++u) {
    qout[i * FD + u] = h[u];
    qout[i * FD + 40 + u] = r[u];
  }
}

// =====================================================================
// packers: f32 [n,80] -> f16 [n,96] (zero pad) and f16 transpose [80,n].
// =====================================================================
__global__ void pack_f16_pad(const float* __restrict__ src, _Float16* __restrict__ dst,
                             int n) {
  int i = blockIdx.x * blockDim.x + threadIdx.x;
  if (i >= n * KP) return;
  int row = i / KP, col = i % KP;
  dst[i] = (col < FD) ? (_Float16)src[row * FD + col] : (_Float16)0.f;
}

__global__ void pack_f16_T(const float* __restrict__ src, _Float16* __restrict__ dst,
                           int n) {
  int i = blockIdx.x * blockDim.x + threadIdx.x;
  if (i >= n * FD) return;
  int row = i / n, col = i % n;               // dst[row, col] = src[col, row]
  dst[i] = (_Float16)src[col * FD + row];
}

// =====================================================================
// interaction[8000,8000] = gu @ gv^T via v_wmma_f32_16x16x32_f16.
// Inputs pre-packed f16 [n,96] (K padded). One wave: one 16-row strip x
// four 16-col tiles; A fragments held in registers across the 4 tiles.
// All fragment loads are contiguous 16B vector loads; no divergence.
// A frag: lane<16 row=m0+lane, elems 0..7 -> K+0..7, 8..15 -> K+16..23;
//         lanes>=16 same rows, K offset +8.
// B frag: lane&15 = output column, same K layout as A frag.
// =====================================================================
__global__ void interact_wmma(const _Float16* __restrict__ guh,
                              const _Float16* __restrict__ gvh,
                              float* __restrict__ Cout, int n) {
  int lane = threadIdx.x & 31;
  int wv = threadIdx.x >> 5;
  int mt = blockIdx.x * 4 + wv;               // 0..499
  int m0 = mt * 16;
  int ntb = blockIdx.y * 4;                   // base n-tile of this block
  bool hi = lane >= 16;
  int lr = lane & 15;
  int offA = hi ? 8 : 0;
  int offB = hi ? 8 : 0;

  const _Float16* arow = guh + (size_t)(m0 + lr) * KP;
  v16h afrag[3];
#pragma unroll
  for (int kc = 0; kc < 3; ++kc) {
    v8h lo = *(const v8h*)(arow + kc * 32 + offA);
    v8h hi8 = *(const v8h*)(arow + kc * 32 + 16 + offA);
    afrag[kc] = cat8(lo, hi8);
  }

#pragma unroll
  for (int t = 0; t < 4; ++t) {
    int n0 = (ntb + t) * 16;
    const _Float16* brow = gvh + (size_t)(n0 + lr) * KP;
    v8f acc = {};
#pragma unroll
    for (int kc = 0; kc < 3; ++kc) {
      v8h lo = *(const v8h*)(brow + kc * 32 + offB);
      v8h hi8 = *(const v8h*)(brow + kc * 32 + offB + 16);
      v16h b = cat8(lo, hi8);
      acc = __builtin_amdgcn_wmma_f32_16x16x32_f16(false, afrag[kc], false, b,
                                                   (short)0, acc, false, false);
    }
#pragma unroll
    for (int r = 0; r < 8; ++r) {
      int row = m0 + (hi ? 8 + r : r);
      Cout[(size_t)row * n + n0 + lr] = acc[r];
    }
  }
}

// =====================================================================
// after = tanh(C) @ W (transp==0) or tanh(C)^T @ W (transp==1).
// WT is the f16-transposed weight [80, n] so B loads are contiguous in K.
// One wave: 16-row strip x all 80 cols (5 accumulators), K loop over n.
// A is read from the f32 interaction (d_out+1, 4B aligned -> scalar b32
// loads) with tanh applied inline; the K chunk 4 ahead is prefetched (transp==0 only).
// =====================================================================
__global__ void after_wmma(const float* __restrict__ Cint, const _Float16* __restrict__ WT,
                           float* __restrict__ out, int n, int transp) {
  int lane = threadIdx.x & 31;
  int wv = threadIdx.x >> 5;
  int mt = blockIdx.x * 4 + wv;
  int m0 = mt * 16;
  bool hi = lane >= 16;
  int lr = lane & 15;
  int offA = hi ? 8 : 0;
  int offB = hi ? 8 : 0;
  v8f acc[5] = {};
  for (int kc = 0; kc < N_NODES / 32; ++kc) {
    int kbase = kc * 32;
    v16h a;
    if (!transp) {
      const float* crow = Cint + (size_t)(m0 + lr) * n + kbase + offA;
      __builtin_prefetch((const void*)(crow + 128), 0, 0);
#pragma unroll
      for (int j = 0; j < 8; ++j) {
        a[j]     = (_Float16)tanhf(crow[j]);
        a[8 + j] = (_Float16)tanhf(crow[16 + j]);
      }
    } else {
      const float* ccol = Cint + (size_t)(kbase + offA) * n + m0 + lr;
#pragma unroll
      for (int j = 0; j < 8; ++j) {
        a[j]     = (_Float16)tanhf(ccol[(size_t)j * n]);
        a[8 + j] = (_Float16)tanhf(ccol[(size_t)(16 + j) * n]);
      }
    }
#pragma unroll
    for (int t = 0; t < 5; ++t) {
      const _Float16* bcol = WT + (size_t)(t * 16 + lr) * n + kbase + offB;
      v8h lo = *(const v8h*)(bcol);
      v8h hi8 = *(const v8h*)(bcol + 16);
      v16h b = cat8(lo, hi8);
      acc[t] = __builtin_amdgcn_wmma_f32_16x16x32_f16(false, a, false, b, (short)0,
                                                      acc[t], false, false);
    }
  }
#pragma unroll
  for (int t = 0; t < 5; ++t) {
#pragma unroll
    for (int r = 0; r < 8; ++r) {
      int row = m0 + (hi ? 8 + r : r);
      out[(size_t)row * FD + t * 16 + lr] = acc[t][r];
    }
  }
}

// =====================================================================
// gather set2set (batch of 1) over all nodes; one block per side (u/v).
// =====================================================================
__global__ void s2s_pool(const float* __restrict__ u_after, const float* __restrict__ gu,
                         const float* __restrict__ v_after, const float* __restrict__ gvv,
                         const float* __restrict__ Wih_u, const float* __restrict__ Whh_u,
                         const float* __restrict__ bih_u, const float* __restrict__ bhh_u,
                         const float* __restrict__ Wih_v, const float* __restrict__ Whh_v,
                         const float* __restrict__ bih_v, const float* __restrict__ bhh_v,
                         float* __restrict__ logits_ws, float* __restrict__ cu,
                         float* __restrict__ cv, int n) {
  __shared__ float h[160], c[160], q[320], g[640], red[256];
  int b = blockIdx.x, tid = threadIdx.x;
  const float* af  = b ? v_after : u_after;
  const float* gg_ = b ? gvv : gu;
  const float* Wih = b ? Wih_v : Wih_u;
  const float* Whh = b ? Whh_v : Whh_u;
  const float* bih = b ? bih_v : bih_u;
  const float* bhh = b ? bhh_v : bhh_u;
  float* lg  = logits_ws + b * n;
  float* dst = b ? cv : cu;
  for (int k = tid; k < 160; k += blockDim.x) { h[k] = 0.f; c[k] = 0.f; }
  for (int k = tid; k < 320; k += blockDim.x) q[k] = 0.f;
  __syncthreads();
  for (int step = 0; step < 2; ++step) {
    for (int j = tid; j < 640; j += blockDim.x) {
      float a = bih[j] + bhh[j];
      const float* wr = &Wih[j * 320];
      const float* hr = &Whh[j * 160];
      for (int c1 = 0; c1 < 320; ++c1) a += wr[c1] * q[c1];
      for (int c1 = 0; c1 < 160; ++c1) a += hr[c1] * h[c1];
      g[j] = a;
    }
    __syncthreads();
    for (int u = tid; u < 160; u += blockDim.x) {
      float ig = sigm(g[u]), fg = sigm(g[160 + u]);
      float gt = tanhf(g[320 + u]), og = sigm(g[480 + u]);
      float cn = fg * c[u] + ig * gt;
      c[u] = cn;
      h[u] = og * tanhf(cn);
    }
    __syncthreads();
    float lmax = -3.4e38f;
    for (int i = tid; i < n; i += blockDim.x) {
      float l = 0.f;
      for (int k = 0; k < 80; ++k) l += af[i * FD + k] * h[k];
      for (int k = 0; k < 80; ++k) l += gg_[i * FD + k] * h[80 + k];
      lg[i] = l;
      lmax = fmaxf(lmax, l);
    }
    red[tid] = lmax; __syncthreads();
    for (int s = blockDim.x / 2; s > 0; s >>= 1) {
      if (tid < s) red[tid] = fmaxf(red[tid], red[tid + s]);
      __syncthreads();
    }
    float m = red[0]; __syncthreads();
    float lsum = 0.f;
    for (int i = tid; i < n; i += blockDim.x) {
      float e = __expf(lg[i] - m);
      lg[i] = e;
      lsum += e;
    }
    red[tid] = lsum; __syncthreads();
    for (int s = blockDim.x / 2; s > 0; s >>= 1) {
      if (tid < s) red[tid] += red[tid + s];
      __syncthreads();
    }
    float ssum = red[0]; __syncthreads();
    for (int k = tid; k < 160; k += blockDim.x) {
      float racc = 0.f;
      if (k < 80) { for (int i = 0; i < n; ++i) racc += lg[i] * af[i * FD + k]; }
      else        { for (int i = 0; i < n; ++i) racc += lg[i] * gg_[i * FD + (k - 80)]; }
      q[160 + k] = racc / ssum;
      q[k] = h[k];
    }
    __syncthreads();
  }
  for (int k = tid; k < 320; k += blockDim.x) dst[k] = q[k];
}

// =====================================================================
// final 4-layer MLP, single block; writes d_out[0].
// =====================================================================
__global__ void mlp_head(const float* __restrict__ cu, const float* __restrict__ cv,
                         const float* __restrict__ w1, const float* __restrict__ b1,
                         const float* __restrict__ w2, const float* __restrict__ b2,
                         const float* __restrict__ w3, const float* __restrict__ b3,
                         const float* __restrict__ w4, const float* __restrict__ b4,
                         float* __restrict__ out) {
  __shared__ float x0[640], x1[360], x2[200], x3[120];
  int tid = threadIdx.x;
  for (int k = tid; k < 640; k += blockDim.x) x0[k] = (k < 320) ? cu[k] : cv[k - 320];
  __syncthreads();
  for (int j = tid; j < 360; j += blockDim.x) {
    float a = b1[j];
    for (int c = 0; c < 640; ++c) a += w1[j * 640 + c] * x0[c];
    x1[j] = fmaxf(a, 0.f);
  }
  __syncthreads();
  for (int j = tid; j < 200; j += blockDim.x) {
    float a = b2[j];
    for (int c = 0; c < 360; ++c) a += w2[j * 360 + c] * x1[c];
    x2[j] = fmaxf(a, 0.f);
  }
  __syncthreads();
  for (int j = tid; j < 120; j += blockDim.x) {
    float a = b3[j];
    for (int c = 0; c < 200; ++c) a += w3[j * 200 + c] * x2[c];
    x3[j] = fmaxf(a, 0.f);
  }
  __syncthreads();
  if (tid == 0) {
    float a = b4[0];
    for (int c = 0; c < 120; ++c) a += w4[c] * x3[c];
    out[0] = a;
  }
}

// =====================================================================
extern "C" void kernel_launch(void* const* d_in, const int* in_sizes, int n_in,
                              void* d_out, int out_size, void* d_ws, size_t ws_size,
                              hipStream_t stream) {
  (void)in_sizes; (void)n_in; (void)out_size; (void)ws_size;
  const float* nodes_u = (const float*)d_in[0];
  const int*   ei_u    = (const int*)d_in[1];
  const float* ef_u    = (const float*)d_in[2];
  const float* nodes_v = (const float*)d_in[3];
  const int*   ei_v    = (const int*)d_in[4];
  const float* ef_v    = (const float*)d_in[5];
  const float* Uw_u = (const float*)d_in[6];
  const float* Ub_u = (const float*)d_in[7];
  const float* Mw_u = (const float*)d_in[8];
  const float* Mb_u = (const float*)d_in[9];
  const float* Uw_v = (const float*)d_in[10];
  const float* Ub_v = (const float*)d_in[11];
  const float* Mw_v = (const float*)d_in[12];
  const float* Mb_v = (const float*)d_in[13];
  const float* ls_Wih = (const float*)d_in[14];
  const float* ls_Whh = (const float*)d_in[15];
  const float* ls_bih = (const float*)d_in[16];
  const float* ls_bhh = (const float*)d_in[17];
  const float* lv_Wih = (const float*)d_in[18];
  const float* lv_Whh = (const float*)d_in[19];
  const float* lv_bih = (const float*)d_in[20];
  const float* lv_bhh = (const float*)d_in[21];
  const float* gs_Wih = (const float*)d_in[22];
  const float* gs_Whh = (const float*)d_in[23];
  const float* gs_bih = (const float*)d_in[24];
  const float* gs_bhh = (const float*)d_in[25];
  const float* gv_Wih = (const float*)d_in[26];
  const float* gv_Whh = (const float*)d_in[27];
  const float* gv_bih = (const float*)d_in[28];
  const float* gv_bhh = (const float*)d_in[29];
  const float* w1 = (const float*)d_in[30];
  const float* b1 = (const float*)d_in[31];
  const float* w2 = (const float*)d_in[32];
  const float* b2 = (const float*)d_in[33];
  const float* w3 = (const float*)d_in[34];
  const float* b3 = (const float*)d_in[35];
  const float* w4 = (const float*)d_in[36];
  const float* b4 = (const float*)d_in[37];

  float* out  = (float*)d_out;
  float* Cint = out + 1;   // interaction, second tuple element, row-major [8000,8000]

  // ---- workspace carve: f16 (aligned) buffers first, then f32 ----
  _Float16* guh = (_Float16*)d_ws;                 // [8000, 96]
  _Float16* gvh = guh + (size_t)N_NODES * KP;      // [8000, 96]
  _Float16* guT = gvh + (size_t)N_NODES * KP;      // [80, 8000]
  _Float16* gvT = guT + (size_t)N_NODES * FD;      // [80, 8000]
  float* w = (float*)(gvT + (size_t)N_NODES * FD);
  float* h_u_a = w; w += N_NODES * ND;
  float* h_u_b = w; w += N_NODES * ND;
  float* msg_u = w; w += N_NODES * ND;
  float* h_v_a = w; w += N_NODES * ND;
  float* h_v_b = w; w += N_NODES * ND;
  float* msg_v = w; w += N_NODES * ND;
  float* gu      = w; w += N_NODES * FD;
  float* gvq     = w; w += N_NODES * FD;
  float* u_after = w; w += N_NODES * FD;
  float* v_after = w; w += N_NODES * FD;
  float* logits  = w; w += 2 * N_NODES;
  float* cu = w; w += 320;
  float* cv = w; w += 320;

  hipMemcpyAsync(h_u_a, nodes_u, N_NODES * ND * sizeof(float),
                 hipMemcpyDeviceToDevice, stream);
  hipMemcpyAsync(h_v_a, nodes_v, N_NODES * ND * sizeof(float),
                 hipMemcpyDeviceToDevice, stream);

  const int EB = (N_EDGES + 255) / 256;
  const int NB = (N_NODES + 255) / 256;

  float* cur_u = h_u_a; float* nxt_u = h_u_b;
  float* cur_v = h_v_a; float* nxt_v = h_v_b;
  for (int k = 0; k < 3; ++k) {
    hipMemsetAsync(msg_u, 0, N_NODES * ND * sizeof(float), stream);
    mp_edge<<<EB, 256, 0, stream>>>(cur_u, ei_u, ef_u, Uw_u + k * ND * 90,
                                    Ub_u + k * ND, msg_u, N_EDGES);
    mp_node<<<NB, 256, 0, stream>>>(cur_u, msg_u, Mw_u + k * ND * 80,
                                    Mb_u + k * ND, nxt_u, N_NODES);
    float* t = cur_u; cur_u = nxt_u; nxt_u = t;

    hipMemsetAsync(msg_v, 0, N_NODES * ND * sizeof(float), stream);
    mp_edge<<<EB, 256, 0, stream>>>(cur_v, ei_v, ef_v, Uw_v + k * ND * 90,
                                    Ub_v + k * ND, msg_v, N_EDGES);
    mp_node<<<NB, 256, 0, stream>>>(cur_v, msg_v, Mw_v + k * ND * 80,
                                    Mb_v + k * ND, nxt_v, N_NODES);
    t = cur_v; cur_v = nxt_v; nxt_v = t;
  }

  s2s_pair<<<(N_NODES + 63) / 64, 64, 0, stream>>>(nodes_u, cur_u, ls_Wih, ls_Whh,
                                                   ls_bih, ls_bhh, gu, N_NODES);
  s2s_pair<<<(N_NODES + 63) / 64, 64, 0, stream>>>(nodes_v, cur_v, lv_Wih, lv_Whh,
                                                   lv_bih, lv_bhh, gvq, N_NODES);

  // pack gu/gv into f16 (K-padded row-major + transposed)
  const int PB1 = (N_NODES * KP + 255) / 256;
  const int PB2 = (N_NODES * FD + 255) / 256;
  pack_f16_pad<<<PB1, 256, 0, stream>>>(gu, guh, N_NODES);
  pack_f16_pad<<<PB1, 256, 0, stream>>>(gvq, gvh, N_NODES);
  pack_f16_T<<<PB2, 256, 0, stream>>>(gu, guT, N_NODES);
  pack_f16_T<<<PB2, 256, 0, stream>>>(gvq, gvT, N_NODES);

  // interaction = gu @ gv^T : 500x500 tiles; wave = 16 rows x 4 col-tiles
  dim3 gInt(N_NODES / 64, N_NODES / 64);           // (125, 125)
  interact_wmma<<<gInt, 128, 0, stream>>>(guh, gvh, Cint, N_NODES);

  // u_after = tanh(C) @ gv ; v_after = tanh(C)^T @ gu
  after_wmma<<<N_NODES / 64, 128, 0, stream>>>(Cint, gvT, u_after, N_NODES, 0);
  after_wmma<<<N_NODES / 64, 128, 0, stream>>>(Cint, guT, v_after, N_NODES, 1);

  s2s_pool<<<2, 256, 0, stream>>>(u_after, gu, v_after, gvq,
                                  gs_Wih, gs_Whh, gs_bih, gs_bhh,
                                  gv_Wih, gv_Whh, gv_bih, gv_bhh,
                                  logits, cu, cv, N_NODES);

  mlp_head<<<1, 384, 0, stream>>>(cu, cv, w1, b1, w2, b2, w3, b3, w4, b4, out);
}
